// Prediction_rep_14559939134020
// MI455X (gfx1250) — compile-verified, matrix operand order checked by reading, not run on hardware
//
#include <hip/hip_runtime.h>
#include <hip/hip_bf16.h>

// ---------------------------------------------------------------------------
// CDNA5 (gfx1250) implementation.
// Batch dimension (B=32768) is the GEMM M dimension; all dense layers run as
// f16 WMMA (v_wmma_f32_16x16x32_f16) with f32 accumulation. The star-graph
// GATs collapse analytically to two distinct rows per example (ego-attended
// row and a shared row), so their Wo projections become [B,48]@[48,64] WMMA
// GEMMs; the attention itself is a 2-way softmax handled per-example.
// Epilogues (activation / bias / output type) are compile-time template
// parameters so the WMMA inner loop and store loop are branch-free.
// ---------------------------------------------------------------------------

typedef _Float16 v16h __attribute__((ext_vector_type(16)));
typedef _Float16 v8h  __attribute__((ext_vector_type(8)));
typedef float    v8f  __attribute__((ext_vector_type(8)));

static __device__ __forceinline__ v16h mk16(v8h lo, v8h hi) {
  v16h r;
#pragma unroll
  for (int i = 0; i < 8; ++i) { r[i] = lo[i]; r[i + 8] = hi[i]; }
  return r;
}

// ---------------------------------------------------------------------------
// Generic WMMA GEMM:  out = act(A[M, KPAD](f16) @ W[K, N](f32) + bias)
// KPAD = KSTEPS*32. A must be zero-padded for k >= K. One 16-row tile per
// wave per grid-stride step; weights transposed into LDS as f16 [N][KPAD].
// A-fragment layout (16-bit A, 16x32): lane<16 -> M=lane, K in {kb..kb+7,
// kb+16..kb+23}; lane>=16 adds koff=8. B assumed transpose-symmetric.
// C/D: vgpr j -> (M = j + 8*(lane>>4), N = lane&15).
// ACT: 0 = none, 1 = relu, 2 = leaky(slope).
// ---------------------------------------------------------------------------
template <int KSTEPS, int ACT, bool HASB, bool OUTF16>
__global__ __launch_bounds__(256) void wmma_gemm_kernel(
    const _Float16* __restrict__ A, int pitchA,
    const float* __restrict__ W, int K, int N,
    const float* __restrict__ bias, float slope,
    _Float16* __restrict__ outH, float* __restrict__ outF,
    int pitchO, int colOff, int M)
{
  constexpr int KPAD = KSTEPS * 32;
  __shared__ _Float16 sW[18432];  // up to N=64 x KPAD=288 (or 128 x 32)

  const int tid = threadIdx.x;
  for (int idx = tid; idx < N * KPAD; idx += 256) {
    const int n = idx / KPAD;
    const int k = idx - n * KPAD;
    sW[idx] = (k < K) ? (_Float16)W[k * N + n] : (_Float16)0.0f;
  }
  __syncthreads();

  const int wave = tid >> 5;
  const int lane = tid & 31;
  const int l16  = lane & 15;
  const int koff = (lane >> 4) * 8;
  const int tilesPerGrid = gridDim.x * 8;

  for (int t = blockIdx.x * 8 + wave; t * 16 < M; t += tilesPerGrid) {
    const int mtile = t * 16;

    v16h afrag[KSTEPS];
    const _Float16* arow = A + (size_t)(mtile + l16) * pitchA + koff;
#pragma unroll
    for (int ks = 0; ks < KSTEPS; ++ks) {
      v8h lo = *(const v8h*)(arow + ks * 32);
      v8h hi = *(const v8h*)(arow + ks * 32 + 16);
      afrag[ks] = mk16(lo, hi);
    }

    for (int nt = 0; nt < N; nt += 16) {
      v8f acc;
#pragma unroll
      for (int j = 0; j < 8; ++j) acc[j] = 0.0f;

      const _Float16* wrow = &sW[(nt + l16) * KPAD + koff];
#pragma unroll
      for (int ks = 0; ks < KSTEPS; ++ks) {
        v8h lo = *(const v8h*)(wrow + ks * 32);
        v8h hi = *(const v8h*)(wrow + ks * 32 + 16);
        v16h bfrag = mk16(lo, hi);
        acc = __builtin_amdgcn_wmma_f32_16x16x32_f16(
            false, afrag[ks], false, bfrag, (short)0, acc, false, false);
      }

      const float bv = HASB ? bias[nt + l16] : 0.0f;
      const size_t base =
          (size_t)(mtile + (lane >> 4) * 8) * pitchO + (size_t)(colOff + nt + l16);
#pragma unroll
      for (int j = 0; j < 8; ++j) {
        float v = acc[j] + bv;
        if (ACT == 1) v = fmaxf(v, 0.0f);
        if (ACT == 2) v = v > 0.0f ? v : v * slope;
        const size_t p = base + (size_t)j * pitchO;
        if (OUTF16) outH[p] = (_Float16)v;
        else        outF[p] = v;
      }
    }
  }
}

// ---------------------------------------------------------------------------
// GAT layer-1 preprocessing (per example). Star graph: rows n>=1 of layer-1
// output all equal elu(h[0]); row 0 is the softmax-attended combination.
// Writes x2_0 -> out[0..47] (pad 48..63 = 0), x2_S -> out[64..111] (pad 0).
// Wh: [3][3][16], a1/a2: [3][16].
// ---------------------------------------------------------------------------
template <int N>
static __device__ __forceinline__ void gat_pre(
    const float x[][3], const float* __restrict__ Wh,
    const float* __restrict__ a1, const float* __restrict__ a2,
    _Float16* __restrict__ out)
{
#pragma unroll
  for (int h = 0; h < 3; ++h) {
    const float* W  = Wh + h * 48;
    const float* A1 = a1 + h * 16;
    const float* A2 = a2 + h * 16;

    float hh0[16];
    float e[N];
    float s1_0 = 0.0f;
#pragma unroll
    for (int m = 0; m < N; ++m) {
      float d2 = 0.0f, d1 = 0.0f;
#pragma unroll
      for (int o = 0; o < 16; ++o) {
        const float t = x[m][0] * W[o] + x[m][1] * W[16 + o] + x[m][2] * W[32 + o];
        d2 += t * A2[o];
        if (m == 0) { d1 += t * A1[o]; hh0[o] = t; }
      }
      e[m] = d2;
      if (m == 0) s1_0 = d1;
    }

    float mx = -1e30f;
#pragma unroll
    for (int m = 0; m < N; ++m) {
      float v = s1_0 + e[m];
      v = v > 0.0f ? v : 0.2f * v;   // leaky 0.2
      e[m] = v;
      mx = fmaxf(mx, v);
    }
    float sum = 0.0f;
#pragma unroll
    for (int m = 0; m < N; ++m) { e[m] = __expf(e[m] - mx); sum += e[m]; }
    const float inv = 1.0f / sum;

    float g[16];
#pragma unroll
    for (int o = 0; o < 16; ++o) g[o] = e[0] * hh0[o];
#pragma unroll
    for (int m = 1; m < N; ++m) {
#pragma unroll
      for (int o = 0; o < 16; ++o) {
        const float t = x[m][0] * W[o] + x[m][1] * W[16 + o] + x[m][2] * W[32 + o];
        g[o] += e[m] * t;
      }
    }
#pragma unroll
    for (int o = 0; o < 16; ++o) {
      const float gv = g[o] * inv;
      const float v0 = gv > 0.0f ? gv : (__expf(gv) - 1.0f);     // elu
      const float hv = hh0[o];
      const float vS = hv > 0.0f ? hv : (__expf(hv) - 1.0f);     // elu
      out[h * 16 + o]      = (_Float16)v0;
      out[64 + h * 16 + o] = (_Float16)vS;
    }
  }
#pragma unroll
  for (int j = 48; j < 64; ++j) { out[j] = (_Float16)0.0f; out[64 + j] = (_Float16)0.0f; }
}

// ---------------------------------------------------------------------------
// Gather / pack kernel: one thread per example.
// ---------------------------------------------------------------------------
__global__ __launch_bounds__(256) void gather_kernel(
    const float* __restrict__ obs,
    const float* __restrict__ hdv_Wh, const float* __restrict__ hdv_a1,
    const float* __restrict__ hdv_a2,
    const float* __restrict__ cav_Wh, const float* __restrict__ cav_a1,
    const float* __restrict__ cav_a2,
    _Float16* __restrict__ A0, _Float16* __restrict__ hdvx2,
    _Float16* __restrict__ cavx2, _Float16* __restrict__ laneIn,
    _Float16* __restrict__ combb, int M)
{
  const int row = blockIdx.x * blockDim.x + threadIdx.x;
  if (row >= M) return;
  const float* o   = obs + (size_t)row * 2064;   // 6 * 344
  const float* cur = o + 1720;                   // step 5

  // ego history: self_[:3] of steps 1..5
  _Float16* a0 = A0 + (size_t)row * 32;
#pragma unroll
  for (int i = 0; i < 5; ++i)
#pragma unroll
    for (int f = 0; f < 3; ++f)
      a0[i * 3 + f] = (_Float16)o[(i + 1) * 344 + 234 + f];
#pragma unroll
  for (int j = 15; j < 32; ++j) a0[j] = (_Float16)0.0f;

  // lane features (108 -> pad 128)
  _Float16* li = laneIn + (size_t)row * 128;
  for (int j = 0; j < 108; ++j) li[j] = (_Float16)cur[120 + j];
#pragma unroll
  for (int j = 108; j < 128; ++j) li[j] = (_Float16)0.0f;

  // comb tail: [road(10) = bott,dist_bott,road_end,dist_end,target][gen][exe], pad
  _Float16* cb = combb + (size_t)row * 288;
  cb[256] = (_Float16)cur[228]; cb[257] = (_Float16)cur[229];  // bott
  cb[258] = (_Float16)cur[247]; cb[259] = (_Float16)cur[248];  // dist_bott
  cb[260] = (_Float16)cur[230]; cb[261] = (_Float16)cur[231];  // road_end
  cb[262] = (_Float16)cur[249]; cb[263] = (_Float16)cur[250];  // dist_end
  cb[264] = (_Float16)cur[232]; cb[265] = (_Float16)cur[233];  // target
  cb[266] = (_Float16)cur[253];                                 // gen
  cb[267] = (_Float16)cur[251]; cb[268] = (_Float16)cur[252];   // exe
#pragma unroll
  for (int j = 269; j < 288; ++j) cb[j] = (_Float16)0.0f;

  // GAT node features: node 0 = ego self_[:3]
  float x[13][3];
  x[0][0] = cur[234]; x[0][1] = cur[235]; x[0][2] = cur[236];
#pragma unroll
  for (int m = 0; m < 12; ++m)
#pragma unroll
    for (int f = 0; f < 3; ++f) x[m + 1][f] = cur[m * 6 + f];
  gat_pre<13>(x, hdv_Wh, hdv_a1, hdv_a2, hdvx2 + (size_t)row * 128);

#pragma unroll
  for (int m = 0; m < 8; ++m)
#pragma unroll
    for (int f = 0; f < 3; ++f) x[m + 1][f] = cur[72 + m * 6 + f];
  gat_pre<9>(x, cav_Wh, cav_a1, cav_a2, cavx2 + (size_t)row * 128);
}

// ---------------------------------------------------------------------------
// GAT layer-2 finish: out = elu(w0*ho0 + wS*hoS) with 2-way softmax where
// the shared row counts (N-1) times. ho: [ho0(64) | hoS(64)] f16.
// ---------------------------------------------------------------------------
static __device__ __forceinline__ void gat_out(
    const _Float16* __restrict__ ho, const float* __restrict__ ao1,
    const float* __restrict__ ao2, int N, _Float16* __restrict__ out)
{
  float p1 = 0.0f, p20 = 0.0f, p2S = 0.0f;
#pragma unroll
  for (int o = 0; o < 64; ++o) {
    const float h0 = (float)ho[o];
    const float hS = (float)ho[64 + o];
    p1  += h0 * ao1[o];
    p20 += h0 * ao2[o];
    p2S += hS * ao2[o];
  }
  float a = p1 + p20; a = a > 0.0f ? a : 0.2f * a;
  float b = p1 + p2S; b = b > 0.0f ? b : 0.2f * b;
  const float mx = fmaxf(a, b);
  const float ea = __expf(a - mx);
  const float eb = __expf(b - mx) * (float)(N - 1);
  const float inv = 1.0f / (ea + eb);
  const float w0 = ea * inv, wS = eb * inv;
#pragma unroll
  for (int o = 0; o < 64; ++o) {
    const float v = w0 * (float)ho[o] + wS * (float)ho[64 + o];
    out[o] = (_Float16)(v > 0.0f ? v : (__expf(v) - 1.0f));   // elu
  }
}

__global__ __launch_bounds__(256) void finish_kernel(
    const _Float16* __restrict__ hdvho, const _Float16* __restrict__ cavho,
    const float* __restrict__ h_ao1, const float* __restrict__ h_ao2,
    const float* __restrict__ c_ao1, const float* __restrict__ c_ao2,
    _Float16* __restrict__ combb, int M)
{
  const int row = blockIdx.x * blockDim.x + threadIdx.x;
  if (row >= M) return;
  gat_out(hdvho + (size_t)row * 128, h_ao1, h_ao2, 13,
          combb + (size_t)row * 288 + 64);
  gat_out(cavho + (size_t)row * 128, c_ao1, c_ao2, 9,
          combb + (size_t)row * 288 + 128);
}

// ---------------------------------------------------------------------------
extern "C" void kernel_launch(void* const* d_in, const int* in_sizes, int n_in,
                              void* d_out, int out_size, void* d_ws, size_t ws_size,
                              hipStream_t stream)
{
  (void)n_in; (void)out_size; (void)ws_size;
  const int M = in_sizes[0] / (6 * 344);

  const float* obs     = (const float*)d_in[0];
  const float* W_self  = (const float*)d_in[1];
  const float* b_self  = (const float*)d_in[2];
  const float* W_dyn   = (const float*)d_in[3];
  const float* b_dyn   = (const float*)d_in[4];
  const float* hdv_Wh  = (const float*)d_in[5];
  const float* hdv_a1  = (const float*)d_in[6];
  const float* hdv_a2  = (const float*)d_in[7];
  const float* hdv_Wo  = (const float*)d_in[8];
  const float* hdv_ao1 = (const float*)d_in[9];
  const float* hdv_ao2 = (const float*)d_in[10];
  const float* cav_Wh  = (const float*)d_in[11];
  const float* cav_a1  = (const float*)d_in[12];
  const float* cav_a2  = (const float*)d_in[13];
  const float* cav_Wo  = (const float*)d_in[14];
  const float* cav_ao1 = (const float*)d_in[15];
  const float* cav_ao2 = (const float*)d_in[16];
  const float* lane_W1 = (const float*)d_in[17];
  const float* lane_b1 = (const float*)d_in[18];
  const float* lane_W2 = (const float*)d_in[19];
  const float* lane_b2 = (const float*)d_in[20];
  const float* comb_W1 = (const float*)d_in[21];
  const float* comb_b1 = (const float*)d_in[22];
  const float* comb_W2 = (const float*)d_in[23];
  const float* comb_b2 = (const float*)d_in[24];

  size_t off = 0;
  auto alloc = [&](size_t bytes) -> char* {
    char* p = (char*)d_ws + off;
    off += (bytes + 255) & ~(size_t)255;
    return p;
  };
  _Float16* A0      = (_Float16*)alloc((size_t)M * 32  * 2);
  _Float16* hid1    = (_Float16*)alloc((size_t)M * 128 * 2);
  _Float16* combb   = (_Float16*)alloc((size_t)M * 288 * 2);
  _Float16* hdvx2   = (_Float16*)alloc((size_t)M * 128 * 2);
  _Float16* hdvho   = (_Float16*)alloc((size_t)M * 128 * 2);
  _Float16* cavx2   = (_Float16*)alloc((size_t)M * 128 * 2);
  _Float16* cavho   = (_Float16*)alloc((size_t)M * 128 * 2);
  _Float16* laneIn  = (_Float16*)alloc((size_t)M * 128 * 2);
  _Float16* laneHid = (_Float16*)alloc((size_t)M * 64  * 2);
  _Float16* combHid = (_Float16*)alloc((size_t)M * 64  * 2);

  const dim3 blk(256);
  const int pblocks = (M + 255) / 256;
  const int tiles   = (M + 15) / 16;
  // ~2 M-tiles per wave: amortizes the per-block weight staging into LDS
  // while still putting 1024 waves on the device.
  int gblocks = (tiles + 15) / 16;
  if (gblocks < 1) gblocks = 1;

  gather_kernel<<<pblocks, blk, 0, stream>>>(
      obs, hdv_Wh, hdv_a1, hdv_a2, cav_Wh, cav_a1, cav_a2,
      A0, hdvx2, cavx2, laneIn, combb, M);

  // ego MLP: leaky(leaky(.,0.1),0.1) folded -> slope 0.01; then linear+bias
  wmma_gemm_kernel<1, 2, true, true><<<gblocks, blk, 0, stream>>>(
      A0, 32, W_self, 15, 128, b_self, 0.01f, hid1, nullptr, 128, 0, M);
  wmma_gemm_kernel<4, 0, true, true><<<gblocks, blk, 0, stream>>>(
      hid1, 128, W_dyn, 128, 64, b_dyn, 0.0f, combb, nullptr, 288, 0, M);

  // GAT Wo projections: ho0 and hoS for hdv and cav
  wmma_gemm_kernel<2, 0, false, true><<<gblocks, blk, 0, stream>>>(
      hdvx2,      128, hdv_Wo, 48, 64, nullptr, 0.0f, hdvho, nullptr, 128, 0,  M);
  wmma_gemm_kernel<2, 0, false, true><<<gblocks, blk, 0, stream>>>(
      hdvx2 + 64, 128, hdv_Wo, 48, 64, nullptr, 0.0f, hdvho, nullptr, 128, 64, M);
  wmma_gemm_kernel<2, 0, false, true><<<gblocks, blk, 0, stream>>>(
      cavx2,      128, cav_Wo, 48, 64, nullptr, 0.0f, cavho, nullptr, 128, 0,  M);
  wmma_gemm_kernel<2, 0, false, true><<<gblocks, blk, 0, stream>>>(
      cavx2 + 64, 128, cav_Wo, 48, 64, nullptr, 0.0f, cavho, nullptr, 128, 64, M);

  // lane MLP (relu x2), second layer writes into comb[:,192:256]
  wmma_gemm_kernel<4, 1, true, true><<<gblocks, blk, 0, stream>>>(
      laneIn, 128, lane_W1, 108, 64, lane_b1, 0.0f, laneHid, nullptr, 64, 0, M);
  wmma_gemm_kernel<2, 1, true, true><<<gblocks, blk, 0, stream>>>(
      laneHid, 64, lane_W2, 64, 64, lane_b2, 0.0f, combb, nullptr, 288, 192, M);

  // GAT layer-2 attention finish -> comb[:,64:192]
  finish_kernel<<<pblocks, blk, 0, stream>>>(
      hdvho, cavho, hdv_ao1, hdv_ao2, cav_ao1, cav_ao2, combb, M);

  // combine MLP (relu x2), final layer stores f32 to d_out [B,64]
  wmma_gemm_kernel<9, 1, true, true><<<gblocks, blk, 0, stream>>>(
      combb, 288, comb_W1, 269, 64, comb_b1, 0.0f, combHid, nullptr, 64, 0, M);
  wmma_gemm_kernel<2, 1, true, false><<<gblocks, blk, 0, stream>>>(
      combHid, 64, comb_W2, 64, 64, comb_b2, 0.0f, nullptr, (float*)d_out, 64, 0, M);
}
